// SelfAttention_51299089383561
// MI455X (gfx1250) — matrix operand order repaired and checked by reading, not compiled, not run on hardware
//
#include <hip/hip_runtime.h>
#include <hip/hip_bf16.h>
#include <cstdint>

// ---------------------------------------------------------------------------
// Self-attention block (GroupNorm -> QKV -> softmax(QK^T/sqrt(d)) V -> proj
// -> +residual) for B=32, C=512, H=W=32 (T=1024 tokens, 1 head, d=512).
//
// GEMMs: v_wmma_f32_16x16x32_f16, 128x128 block tile, 8 waves, K-step 64,
// double-buffered LDS staged purely with global_load_async_to_lds_b128
// (ASYNCcnt-tracked). Weights pre-converted to f16 once.
//
// Workspace layout (260 MB total):
//   [  0MB) normT f16 [B,T,C]   (reused later as O f16 [B,T,C])
//   [ 32MB) Q     f16 [B,T,D]   (pre-scaled by 1/sqrt(512))
//   [ 64MB) K     f16 [B,T,D]
//   [ 96MB) Vt    f16 [B,D,T]
//   [128MB) S     f32 [B,T,T]   (softmax writes P f16 in place, row pitch 2048)
//   [256MB) wq16  f16 [1536,512], wo16 f16 [512,512]
// ---------------------------------------------------------------------------

typedef _Float16 v4h  __attribute__((ext_vector_type(4)));
typedef _Float16 v8h  __attribute__((ext_vector_type(8)));
typedef _Float16 v16h __attribute__((ext_vector_type(16)));
typedef float    v4f  __attribute__((ext_vector_type(4)));
typedef float    v8f  __attribute__((ext_vector_type(8)));

#define TOKENS    1024
#define CCH       512
#define BATCH     32
#define KSTEP     64
#define LDS_PITCH 72   // halfs per LDS row (144B: 16B-aligned, bank-friendly)

// ---- CDNA5 async global->LDS copy (ASYNCcnt) ------------------------------
__device__ __forceinline__ void async_load_b128(uint32_t lds_off, const void* gptr)
{
    asm volatile("global_load_async_to_lds_b128 %0, %1, off"
                 :: "v"(lds_off), "v"((uint64_t)(uintptr_t)gptr)
                 : "memory");
}
__device__ __forceinline__ void wait_async0()
{
    asm volatile("s_wait_asynccnt 0x0" ::: "memory");
}
// Generic LDS pointer -> wave-relative LDS byte offset (low 32 bits of the
// LDS-aperture flat address hold the offset).
__device__ __forceinline__ uint32_t lds_off(const void* p)
{
    return (uint32_t)(uintptr_t)p;
}

// ---------------------------------------------------------------------------
// Stage one 128x64 A tile and one 128x64 B tile (f16) into LDS buffer `buf`.
// 256 threads: each issues 4+4 async b128 copies (8 halfs each).
// ---------------------------------------------------------------------------
__device__ __forceinline__ void stage_tiles(
    int buf, int k0,
    const _Float16* __restrict__ Ag, int lda,
    const _Float16* __restrict__ Bg, int ldb,
    _Float16 (&sA)[2][128 * LDS_PITCH], _Float16 (&sB)[2][128 * LDS_PITCH],
    int tid)
{
    const int r = tid >> 3;            // 0..31
    const int c = (tid & 7) * 8;       // 0..56
    #pragma unroll
    for (int p = 0; p < 4; ++p) {
        const int row = r + p * 32;
        async_load_b128(lds_off(&sA[buf][row * LDS_PITCH + c]),
                        Ag + (size_t)row * lda + k0 + c);
        async_load_b128(lds_off(&sB[buf][row * LDS_PITCH + c]),
                        Bg + (size_t)row * ldb + k0 + c);
    }
}

// ---------------------------------------------------------------------------
// Tiled GEMM core: D(128x128 f32) = A(128xK f16) * B^T(128xK), B stored
// row-per-output-column (ldb along K). 256 threads = 8 waves (4 in M, 2 in N),
// each wave owns a 32x64 tile = 2x4 WMMA accumulators. Double-buffered LDS,
// 2 WMMA K-substeps (2x32) per staged 64-deep tile.
// ---------------------------------------------------------------------------
__device__ __forceinline__ void gemm_core(const _Float16* __restrict__ Ag, int lda,
                                          const _Float16* __restrict__ Bg, int ldb,
                                          int kdim, v8f (&acc)[2][4])
{
    __shared__ __align__(16) _Float16 sA[2][128 * LDS_PITCH];
    __shared__ __align__(16) _Float16 sB[2][128 * LDS_PITCH];

    const int tid    = threadIdx.x;
    const int lane   = tid & 31;
    const int wave   = tid >> 5;
    const int half   = lane >> 4;
    const int ln     = lane & 15;
    const int waveMi = (wave & 3) * 32;
    const int waveNi = (wave >> 2) * 64;

    v8f z = {};
    #pragma unroll
    for (int ti = 0; ti < 2; ++ti)
        #pragma unroll
        for (int tj = 0; tj < 4; ++tj) acc[ti][tj] = z;

    stage_tiles(0, 0, Ag, lda, Bg, ldb, sA, sB, tid);
    wait_async0();
    __syncthreads();

    const int nk = kdim / KSTEP;
    for (int i = 0; i < nk; ++i) {
        const int cur = i & 1, nxt = cur ^ 1;

        // overlap: prefetch next K-tile into the other buffer first
        if (i + 1 < nk)
            stage_tiles(nxt, (i + 1) * KSTEP, Ag, lda, Bg, ldb, sA, sB, tid);

        #pragma unroll
        for (int kk = 0; kk < KSTEP; kk += 32) {
            // fragments (ISA 16-bit A layout / same K order for B)
            v16h afrag[2], bfrag[4];
            #pragma unroll
            for (int ti = 0; ti < 2; ++ti) {
                const _Float16* ap = &sA[cur][(waveMi + 16 * ti + ln) * LDS_PITCH + kk];
                v8h lo = *(const v8h*)(ap + half * 8);       // K 0..7  / 8..15
                v8h hi = *(const v8h*)(ap + 16 + half * 8);  // K 16..23 / 24..31
                afrag[ti] = __builtin_shufflevector(lo, hi, 0,1,2,3,4,5,6,7,8,9,10,11,12,13,14,15);
            }
            #pragma unroll
            for (int tj = 0; tj < 4; ++tj) {
                const _Float16* bp = &sB[cur][(waveNi + 16 * tj + ln) * LDS_PITCH + kk];
                v8h lo = *(const v8h*)(bp + half * 8);       // K 0..7  / 8..15
                v8h hi = *(const v8h*)(bp + 16 + half * 8);  // K 16..23 / 24..31
                bfrag[tj] = __builtin_shufflevector(lo, hi, 0,1,2,3,4,5,6,7,8,9,10,11,12,13,14,15);
            }
            #pragma unroll
            for (int ti = 0; ti < 2; ++ti)
                #pragma unroll
                for (int tj = 0; tj < 4; ++tj)
                    acc[ti][tj] = __builtin_amdgcn_wmma_f32_16x16x32_f16(
                        false, afrag[ti], false, bfrag[tj], (short)0, acc[ti][tj], false, false);
        }

        wait_async0();
        __syncthreads();
    }
}

// ---------------------------------------------------------------------------
// Kernel 0: f32 -> f16 elementwise convert (weights, done once)
// ---------------------------------------------------------------------------
__global__ __launch_bounds__(256) void cvt_f16_kernel(
    const float* __restrict__ in, _Float16* __restrict__ out)
{
    const int i = (blockIdx.x * 256 + threadIdx.x) * 4;
    v4f v = *(const v4f*)(in + i);
    v4h h;
    #pragma unroll
    for (int e = 0; e < 4; ++e) h[e] = (_Float16)v[e];
    *(v4h*)(out + i) = h;
}

// ---------------------------------------------------------------------------
// Kernel 1: GroupNorm(32 groups) + transpose to token-major f16 normT[b,t,c]
// ---------------------------------------------------------------------------
__global__ __launch_bounds__(256) void groupnorm_kernel(
    const float* __restrict__ x, const float* __restrict__ gamma,
    const float* __restrict__ beta, _Float16* __restrict__ normT)
{
    const int g = blockIdx.x, b = blockIdx.y, tid = threadIdx.x;
    const float* xg = x + ((size_t)b * CCH + g * 16) * TOKENS;  // 16 chans contiguous

    float s = 0.f, ss = 0.f;
    for (int i = tid; i < 16 * TOKENS; i += 256) { float v = xg[i]; s += v; ss += v * v; }

    __shared__ float rs[256], rss[256];
    rs[tid] = s; rss[tid] = ss; __syncthreads();
    for (int st = 128; st > 0; st >>= 1) {
        if (tid < st) { rs[tid] += rs[tid + st]; rss[tid] += rss[tid + st]; }
        __syncthreads();
    }
    __shared__ float mean_s, inv_s;
    if (tid == 0) {
        float m = rs[0] * (1.f / 16384.f);
        float v = rss[0] * (1.f / 16384.f) - m * m;
        mean_s = m; inv_s = rsqrtf(v + 1e-5f);
    }
    __syncthreads();
    const float mean = mean_s, inv = inv_s;

    __shared__ float tile[16][65];
    for (int t0 = 0; t0 < TOKENS; t0 += 64) {
        #pragma unroll
        for (int q = 0; q < 4; ++q) {           // coalesced read: 16c x 64t
            int lin = q * 256 + tid;
            int c = lin >> 6, tl = lin & 63;
            tile[c][tl] = xg[(size_t)c * TOKENS + t0 + tl];
        }
        __syncthreads();
        #pragma unroll
        for (int q = 0; q < 4; ++q) {           // coalesced write: 64t x 16c
            int lin = q * 256 + tid;
            int tl = lin >> 4, c = lin & 15;
            float gm = gamma[g * 16 + c], bt = beta[g * 16 + c];
            float v = (tile[c][tl] - mean) * inv * gm + bt;
            normT[((size_t)b * TOKENS + t0 + tl) * CCH + g * 16 + c] = (_Float16)v;
        }
        __syncthreads();
    }
}

// ---------------------------------------------------------------------------
// Kernel 2: QKV GEMM  qkv[t,o] = normT[t,:]·wq16[o,:] + b_qkv[o]
//           Q scaled by 1/sqrt(512); V stored transposed [c,t].
//           Region switch is uniform per block (boundaries 512,1024 % 128 == 0).
// ---------------------------------------------------------------------------
__global__ __launch_bounds__(256) void qkv_gemm_kernel(
    const _Float16* __restrict__ normT, const _Float16* __restrict__ wq16,
    const float* __restrict__ b_qkv, _Float16* __restrict__ Q,
    _Float16* __restrict__ Kb, _Float16* __restrict__ Vt)
{
    const int b = blockIdx.z, tileM = blockIdx.x * 128, tileN = blockIdx.y * 128;
    const _Float16* Ag = normT + ((size_t)b * TOKENS + tileM) * CCH;
    const _Float16* Bg = wq16 + (size_t)tileN * CCH;

    v8f acc[2][4];
    gemm_core(Ag, CCH, Bg, CCH, CCH, acc);

    const int lane = threadIdx.x & 31, wave = threadIdx.x >> 5;
    const int half = lane >> 4, ln = lane & 15;
    const int waveMi = (wave & 3) * 32, waveNi = (wave >> 2) * 64;
    const float qscale = 0.044194173824159216f;  // 1/sqrt(512)

    if (tileN < CCH) {                       // Q region (uniform for block)
        #pragma unroll
        for (int ti = 0; ti < 2; ++ti)
            #pragma unroll
            for (int tj = 0; tj < 4; ++tj)
                #pragma unroll
                for (int r = 0; r < 8; ++r) {
                    int m = tileM + waveMi + 16 * ti + r + 8 * half;
                    int n = tileN + waveNi + 16 * tj + ln;
                    float val = acc[ti][tj][r] + b_qkv[n];
                    Q[((size_t)b * TOKENS + m) * CCH + n] = (_Float16)(val * qscale);
                }
    } else if (tileN < 2 * CCH) {            // K region
        #pragma unroll
        for (int ti = 0; ti < 2; ++ti)
            #pragma unroll
            for (int tj = 0; tj < 4; ++tj)
                #pragma unroll
                for (int r = 0; r < 8; ++r) {
                    int m = tileM + waveMi + 16 * ti + r + 8 * half;
                    int n = tileN + waveNi + 16 * tj + ln;
                    float val = acc[ti][tj][r] + b_qkv[n];
                    Kb[((size_t)b * TOKENS + m) * CCH + (n - CCH)] = (_Float16)val;
                }
    } else {                                 // V region (store transposed)
        #pragma unroll
        for (int ti = 0; ti < 2; ++ti)
            #pragma unroll
            for (int tj = 0; tj < 4; ++tj)
                #pragma unroll
                for (int r = 0; r < 8; ++r) {
                    int m = tileM + waveMi + 16 * ti + r + 8 * half;
                    int n = tileN + waveNi + 16 * tj + ln;
                    float val = acc[ti][tj][r] + b_qkv[n];
                    Vt[((size_t)b * CCH + (n - 2 * CCH)) * TOKENS + m] = (_Float16)val;
                }
    }
}

// ---------------------------------------------------------------------------
// Kernel 3: scores S[t1,t2] = Q[t1,:]·K[t2,:]  (f32 out)
// ---------------------------------------------------------------------------
__global__ __launch_bounds__(256) void scores_gemm_kernel(
    const _Float16* __restrict__ Q, const _Float16* __restrict__ Kb,
    float* __restrict__ S)
{
    const int b = blockIdx.z, tileM = blockIdx.x * 128, tileN = blockIdx.y * 128;
    const _Float16* Ag = Q  + ((size_t)b * TOKENS + tileM) * CCH;
    const _Float16* Bg = Kb + ((size_t)b * TOKENS + tileN) * CCH;

    v8f acc[2][4];
    gemm_core(Ag, CCH, Bg, CCH, CCH, acc);

    const int lane = threadIdx.x & 31, wave = threadIdx.x >> 5;
    const int half = lane >> 4, ln = lane & 15;
    const int waveMi = (wave & 3) * 32, waveNi = (wave >> 2) * 64;

    #pragma unroll
    for (int ti = 0; ti < 2; ++ti)
        #pragma unroll
        for (int tj = 0; tj < 4; ++tj)
            #pragma unroll
            for (int r = 0; r < 8; ++r) {
                int m = tileM + waveMi + 16 * ti + r + 8 * half;
                int n = tileN + waveNi + 16 * tj + ln;
                S[((size_t)b * TOKENS + m) * TOKENS + n] = acc[ti][tj][r];
            }
}

// ---------------------------------------------------------------------------
// Kernel 4: row softmax over 1024 (f32 in, f16 out written in place into the
//           same 4KB row region; P row pitch = 2048 halfs)
// ---------------------------------------------------------------------------
__global__ __launch_bounds__(256) void softmax_kernel(float* __restrict__ S)
{
    const int t1 = blockIdx.x, b = blockIdx.y, tid = threadIdx.x;
    float*     srow = S + ((size_t)b * TOKENS + t1) * TOKENS;
    _Float16*  prow = (_Float16*)S + ((size_t)b * TOKENS + t1) * (2 * TOKENS);

    float v[4];
    #pragma unroll
    for (int q = 0; q < 4; ++q) v[q] = srow[q * 256 + tid];

    __shared__ float red[256];
    float mx = fmaxf(fmaxf(v[0], v[1]), fmaxf(v[2], v[3]));
    red[tid] = mx; __syncthreads();
    for (int st = 128; st > 0; st >>= 1) {
        if (tid < st) red[tid] = fmaxf(red[tid], red[tid + st]);
        __syncthreads();
    }
    mx = red[0]; __syncthreads();

    float e[4], s = 0.f;
    #pragma unroll
    for (int q = 0; q < 4; ++q) { e[q] = __expf(v[q] - mx); s += e[q]; }
    red[tid] = s; __syncthreads();
    for (int st = 128; st > 0; st >>= 1) {
        if (tid < st) red[tid] += red[tid + st];
        __syncthreads();
    }
    const float rinv = 1.f / red[0];
    __syncthreads();   // all reads of srow/red done before aliased writes

    #pragma unroll
    for (int q = 0; q < 4; ++q) prow[q * 256 + tid] = (_Float16)(e[q] * rinv);
}

// ---------------------------------------------------------------------------
// Kernel 5: O[t1,c] = P[t1,:]·Vt[c,:]   (f16 out, into reused normT buffer)
// ---------------------------------------------------------------------------
__global__ __launch_bounds__(256) void pv_gemm_kernel(
    const _Float16* __restrict__ P, const _Float16* __restrict__ Vt,
    _Float16* __restrict__ O)
{
    const int b = blockIdx.z, tileM = blockIdx.x * 128, tileN = blockIdx.y * 128;
    const _Float16* Ag = P  + ((size_t)b * TOKENS + tileM) * (2 * TOKENS);
    const _Float16* Bg = Vt + ((size_t)b * CCH + tileN) * TOKENS;

    v8f acc[2][4];
    gemm_core(Ag, 2 * TOKENS, Bg, TOKENS, TOKENS, acc);

    const int lane = threadIdx.x & 31, wave = threadIdx.x >> 5;
    const int half = lane >> 4, ln = lane & 15;
    const int waveMi = (wave & 3) * 32, waveNi = (wave >> 2) * 64;

    #pragma unroll
    for (int ti = 0; ti < 2; ++ti)
        #pragma unroll
        for (int tj = 0; tj < 4; ++tj)
            #pragma unroll
            for (int r = 0; r < 8; ++r) {
                int m = tileM + waveMi + 16 * ti + r + 8 * half;
                int n = tileN + waveNi + 16 * tj + ln;
                O[((size_t)b * TOKENS + m) * CCH + n] = (_Float16)acc[ti][tj][r];
            }
}

// ---------------------------------------------------------------------------
// Kernel 6: out[b,o,t] = O[t,:]·wo16[o,:] + b_out[o] + x[b,o,t]
// ---------------------------------------------------------------------------
__global__ __launch_bounds__(256) void proj_gemm_kernel(
    const _Float16* __restrict__ O, const _Float16* __restrict__ wo16,
    const float* __restrict__ b_out, const float* __restrict__ x,
    float* __restrict__ out)
{
    const int b = blockIdx.z, tileM = blockIdx.x * 128, tileN = blockIdx.y * 128;
    const _Float16* Ag = O + ((size_t)b * TOKENS + tileM) * CCH;
    const _Float16* Bg = wo16 + (size_t)tileN * CCH;

    v8f acc[2][4];
    gemm_core(Ag, CCH, Bg, CCH, CCH, acc);

    const int lane = threadIdx.x & 31, wave = threadIdx.x >> 5;
    const int half = lane >> 4, ln = lane & 15;
    const int waveMi = (wave & 3) * 32, waveNi = (wave >> 2) * 64;

    #pragma unroll
    for (int ti = 0; ti < 2; ++ti)
        #pragma unroll
        for (int tj = 0; tj < 4; ++tj)
            #pragma unroll
            for (int r = 0; r < 8; ++r) {
                int m = tileM + waveMi + 16 * ti + r + 8 * half;   // token
                int n = tileN + waveNi + 16 * tj + ln;             // out channel
                size_t idx = ((size_t)b * CCH + n) * TOKENS + m;
                out[idx] = acc[ti][tj][r] + b_out[n] + x[idx];
            }
}

// ---------------------------------------------------------------------------
extern "C" void kernel_launch(void* const* d_in, const int* in_sizes, int n_in,
                              void* d_out, int out_size, void* d_ws, size_t ws_size,
                              hipStream_t stream)
{
    const float* x      = (const float*)d_in[0];
    const float* gamma  = (const float*)d_in[1];
    const float* beta   = (const float*)d_in[2];
    const float* w_qkv  = (const float*)d_in[3];
    const float* b_qkv  = (const float*)d_in[4];
    const float* w_out  = (const float*)d_in[5];
    const float* b_out  = (const float*)d_in[6];
    float* out = (float*)d_out;

    char* ws = (char*)d_ws;
    _Float16* normT = (_Float16*)(ws);                     // 32 MB (reused as O)
    _Float16* Q     = (_Float16*)(ws + (32ull << 20));     // 32 MB
    _Float16* Kb    = (_Float16*)(ws + (64ull << 20));     // 32 MB
    _Float16* Vt    = (_Float16*)(ws + (96ull << 20));     // 32 MB
    float*    S     = (float*)   (ws + (128ull << 20));    // 128 MB (P in place)
    _Float16* wq16  = (_Float16*)(ws + (256ull << 20));    // 1.5 MB
    _Float16* wo16  = (_Float16*)(ws + (258ull << 20));    // 0.5 MB
    _Float16* O     = normT;                               // lifetime-disjoint reuse

    // 0) one-time weight conversion to f16
    cvt_f16_kernel<<<dim3((3 * CCH * CCH) / 1024), 256, 0, stream>>>(w_qkv, wq16);
    cvt_f16_kernel<<<dim3((CCH * CCH) / 1024), 256, 0, stream>>>(w_out, wo16);
    // 1) GroupNorm -> f16 token-major
    groupnorm_kernel<<<dim3(32, BATCH), 256, 0, stream>>>(x, gamma, beta, normT);
    // 2) QKV GEMM (+bias, Q pre-scaled, V transposed)
    qkv_gemm_kernel<<<dim3(TOKENS / 128, (3 * CCH) / 128, BATCH), 256, 0, stream>>>(
        normT, wq16, b_qkv, Q, Kb, Vt);
    // 3) scores = Q K^T
    scores_gemm_kernel<<<dim3(TOKENS / 128, TOKENS / 128, BATCH), 256, 0, stream>>>(Q, Kb, S);
    // 4) softmax rows -> f16 P in place
    softmax_kernel<<<dim3(TOKENS, BATCH), 256, 0, stream>>>(S);
    // 5) O = P V
    pv_gemm_kernel<<<dim3(TOKENS / 128, CCH / 128, BATCH), 256, 0, stream>>>(
        (const _Float16*)S, Vt, O);
    // 6) proj + bias + residual
    proj_gemm_kernel<<<dim3(TOKENS / 128, CCH / 128, BATCH), 256, 0, stream>>>(
        O, wo16, b_out, x, out);
}
